// AttentionLayer_32521492366094
// MI455X (gfx1250) — matrix operand order repaired and checked by reading, not compiled, not run on hardware
//
#include <hip/hip_runtime.h>

#define BB 8
#define CC 256   // channels (N in reference)
#define MM 64    // head dim (M)
#define LL 4096  // sequence = H*W
#define OO 384   // 2*M + N rows of Wqkv

typedef _Float16 v16h __attribute__((ext_vector_type(16)));
typedef _Float16 v8h  __attribute__((ext_vector_type(8)));
typedef _Float16 h2   __attribute__((ext_vector_type(2)));
typedef float    v8f  __attribute__((ext_vector_type(8)));
typedef int      vi8  __attribute__((ext_vector_type(8)));

static __device__ __forceinline__ v16h cat8(v8h lo, v8h hi) {
  return __builtin_shufflevector(lo, hi, 0,1,2,3,4,5,6,7,8,9,10,11,12,13,14,15);
}
// lane <-> lane^16 exchange (wave32): ds_swizzle_b32 SWAPX16 = offset 0x401f
static __device__ __forceinline__ float swzx16f(float v) {
  return __int_as_float(__builtin_amdgcn_ds_swizzle(__float_as_int(v), 0x401f));
}
static __device__ __forceinline__ int swzx16i(int v) {
  return __builtin_amdgcn_ds_swizzle(v, 0x401f);
}
static __device__ __forceinline__ int packh2(float a, float b) {
  h2 h; h.x = (_Float16)a; h.y = (_Float16)b;
  return __builtin_bit_cast(int, h);
}
// gfx1250 async global->LDS copy (ASYNCcnt tracked), 16 bytes per op.
static __device__ __forceinline__ void async_copy_b128(unsigned lds_addr,
                                                       unsigned long long gaddr) {
  asm volatile("global_load_async_to_lds_b128 %0, %1, off"
               :: "v"(lds_addr), "v"(gaddr) : "memory");
}

// ---------------------------------------------------------------------------
// Kernel 1: qkv = Wqkv @ relu(x) + b  (per batch GEMM 384x256 * 256x4096)
// Outputs (f16 workspace):
//   qt[b][l][64] : q transposed, pre-scaled by (1/sqrt(64))*log2(e)
//   kt[b][l][64] : k transposed
//   vv[b][c][4096] : v
// ---------------------------------------------------------------------------
__global__ __launch_bounds__(256)
void qkv_relu_gemm(const float* __restrict__ x, const float* __restrict__ W,
                   const float* __restrict__ bias,
                   _Float16* __restrict__ qt, _Float16* __restrict__ kt,
                   _Float16* __restrict__ vv)
{
  __shared__ _Float16 xs[32 * 64];   // [c_local][l_local], 4 KB

  const int tid  = threadIdx.x;
  const int lane = tid & 31;
  const int wave = tid >> 5;
  const bool hi  = (lane & 16) != 0;
  const int l16  = lane & 15;

  int blk = blockIdx.x;
  const int lblk = blk & 63;  blk >>= 6;
  const int oblk = blk % 12;
  const int b    = blk / 12;

  const int obase = (oblk * 2 + (wave >> 2)) * 16;   // 0..368, step 16
  const int lloc  = (wave & 3) * 16 + l16;           // 0..63 inside block tile
  const int lbase = lblk * 64;
  const int l     = lbase + lloc;

  v8f acc = {0.f,0.f,0.f,0.f,0.f,0.f,0.f,0.f};

  const float* wrow = W + (size_t)(obase + l16) * CC + (hi ? 8 : 0);

  for (int kc = 0; kc < CC; kc += 32) {
    // stage relu(x)[kc..kc+31][lbase..lbase+63] -> f16 LDS (coalesced rows)
    #pragma unroll
    for (int s = 0; s < 8; ++s) {
      int flat = s * 256 + tid;
      int cl = flat >> 6, ll = flat & 63;
      float vx = x[((size_t)b * CC + (kc + cl)) * LL + lbase + ll];
      xs[flat] = (_Float16)fmaxf(vx, 0.0f);
    }
    __syncthreads();

    // A operand (16x32 f16): lo lanes K=kc+0..7 / +16..23, hi lanes +8 / +24
    v16h a;
    #pragma unroll
    for (int e = 0; e < 8; ++e) {
      a[e]     = (_Float16)wrow[kc + e];
      a[e + 8] = (_Float16)wrow[kc + 16 + e];
    }
    // B operand (32x16 f16): lo lanes K=0..7 / 16..23, hi lanes K=8..15 / 24..31
    v16h bm;
    #pragma unroll
    for (int e = 0; e < 16; ++e)
      bm[e] = xs[((e & 8) * 2 + (hi ? 8 : 0) + (e & 7)) * 64 + lloc];

    acc = __builtin_amdgcn_wmma_f32_16x16x32_f16(false, a, false, bm,
                                                 (short)0, acc, false, false);
    __syncthreads();
  }

  // Epilogue: bias, route to q^T / k^T / v
  const int obias = obase + (hi ? 8 : 0);
  if (obase < MM) {                       // q -> qt[b][l][m], folded scale
    const float sc = 0.125f * 1.44269504088896f;   // 1/sqrt(64) * log2(e)
    v8h p;
    #pragma unroll
    for (int r = 0; r < 8; ++r)
      p[r] = (_Float16)((acc[r] + bias[obias + r]) * sc);
    *(v8h*)(qt + ((size_t)b * LL + l) * MM + obias) = p;
  } else if (obase < 2 * MM) {            // k -> kt[b][l][m]
    v8h p;
    #pragma unroll
    for (int r = 0; r < 8; ++r)
      p[r] = (_Float16)(acc[r] + bias[obias + r]);
    *(v8h*)(kt + ((size_t)b * LL + l) * MM + (obias - MM)) = p;
  } else {                                // v -> vv[b][c][l]
    #pragma unroll
    for (int r = 0; r < 8; ++r) {
      int c = obias - 2 * MM + r;
      vv[((size_t)b * CC + c) * LL + l] = (_Float16)(acc[r] + bias[obias + r]);
    }
  }
}

// ---------------------------------------------------------------------------
// Kernel 2: flash attention with async-LDS double buffering of the V tile.
//   S'[i,j] = sum_m qt[i][m] * kt[j][m]   (already scaled, exp2 domain)
//   p = softmax_i(S'), O[c,j] = sum_i v[c,i] p[i,j]
//   out = x + gamma * O / l
// One wave owns 16 j columns and all 256 channels (16 v8f accumulators).
// All 8 waves share each 256x32 V tile, staged LDS via async b128 copies.
// ---------------------------------------------------------------------------
__global__ __launch_bounds__(256)
void flash_attn(const _Float16* __restrict__ qt, const _Float16* __restrict__ kt,
                const _Float16* __restrict__ vv, const float* __restrict__ x,
                const float* __restrict__ gamma, float* __restrict__ out)
{
  __shared__ _Float16 vs[2][CC * 32];    // double-buffered V tile, 2 x 16 KB

  const int tid  = threadIdx.x;
  const int lane = tid & 31;
  const int wave = tid >> 5;
  const bool hi  = (lane & 16) != 0;
  const int l16  = lane & 15;

  const int b = blockIdx.x >> 5;
  const int j = (blockIdx.x & 31) * 128 + wave * 16 + l16;

  // B operand for scores: K columns (contraction dim m=64 -> two chunks)
  const _Float16* krow = kt + ((size_t)b * LL + j) * MM + (hi ? 8 : 0);
  const v16h bk0 = cat8(*(const v8h*)(krow),      *(const v8h*)(krow + 16));
  const v16h bk1 = cat8(*(const v8h*)(krow + 32), *(const v8h*)(krow + 48));

  const v8f vzero = {0.f,0.f,0.f,0.f,0.f,0.f,0.f,0.f};
  v8f acc[16];
  #pragma unroll
  for (int t = 0; t < 16; ++t) acc[t] = vzero;

  float mrun = -1.0e30f;
  float lrun = 0.0f;

  const _Float16* qtb = qt + (size_t)b * LL * MM;
  // thread t owns V row c = t for staging
  const _Float16* vrow_g = vv + ((size_t)b * CC + tid) * LL;
  const int moff = hi ? 8 : 0;

  // issue 4 async b128 copies: V[tid][ii..ii+31] -> vs[buf][tid][0..31]
  auto stage_v = [&](int ii, int buf) {
    #pragma unroll
    for (int k = 0; k < 4; ++k) {
      unsigned la = (unsigned)(uintptr_t)&vs[buf][tid * 32 + k * 8];
      unsigned long long ga = (unsigned long long)(uintptr_t)(vrow_g + ii + k * 8);
      async_copy_b128(la, ga);
    }
  };

  stage_v(0, 0);   // prologue: tile 0 in flight

  for (int it = 0; it < LL / 32; ++it) {
    const int i0 = it * 32;
    __syncthreads();   // everyone done reading the buffer we are about to fill
    const int inext = (i0 + 32 < LL) ? (i0 + 32) : 0;
    stage_v(inext, (it + 1) & 1);
    // previous 4 copies (current tile) complete when <= 4 remain outstanding
    asm volatile("s_wait_asynccnt 0x4" ::: "memory");
    __syncthreads();   // current tile resident for all waves

    // ---- scores for two 16-row i sub-tiles (A = q^T, 16x32 f16 layout) ----
    const _Float16* q0 = qtb + (size_t)(i0 + l16) * MM + moff;
    const _Float16* q1 = qtb + (size_t)(i0 + 16 + l16) * MM + moff;
    v16h aq00 = cat8(*(const v8h*)(q0),      *(const v8h*)(q0 + 16));
    v16h aq01 = cat8(*(const v8h*)(q0 + 32), *(const v8h*)(q0 + 48));
    v16h aq10 = cat8(*(const v8h*)(q1),      *(const v8h*)(q1 + 16));
    v16h aq11 = cat8(*(const v8h*)(q1 + 32), *(const v8h*)(q1 + 48));

    v8f s1 = vzero, s2 = vzero;
    s1 = __builtin_amdgcn_wmma_f32_16x16x32_f16(false, aq00, false, bk0, (short)0, s1, false, false);
    s1 = __builtin_amdgcn_wmma_f32_16x16x32_f16(false, aq01, false, bk1, (short)0, s1, false, false);
    s2 = __builtin_amdgcn_wmma_f32_16x16x32_f16(false, aq10, false, bk0, (short)0, s2, false, false);
    s2 = __builtin_amdgcn_wmma_f32_16x16x32_f16(false, aq11, false, bk1, (short)0, s2, false, false);

    // ---- online softmax over i (column j split across lane and lane^16) ----
    float mx = s1[0];
    #pragma unroll
    for (int r = 1; r < 8; ++r) mx = fmaxf(mx, s1[r]);
    #pragma unroll
    for (int r = 0; r < 8; ++r) mx = fmaxf(mx, s2[r]);
    mx = fmaxf(mx, swzx16f(mx));
    const float mnew = fmaxf(mrun, mx);
    const float corr = __builtin_amdgcn_exp2f(mrun - mnew);

    float p1[8], p2[8], rs = 0.0f;
    #pragma unroll
    for (int r = 0; r < 8; ++r) { p1[r] = __builtin_amdgcn_exp2f(s1[r] - mnew); rs += p1[r]; }
    #pragma unroll
    for (int r = 0; r < 8; ++r) { p2[r] = __builtin_amdgcn_exp2f(s2[r] - mnew); rs += p2[r]; }
    lrun = lrun * corr + (rs + swzx16f(rs));
    mrun = mnew;

    #pragma unroll
    for (int t = 0; t < 16; ++t)
      #pragma unroll
      for (int r = 0; r < 8; ++r)
        acc[t][r] *= corr;

    // ---- repack P (f32, D layout) into f16 B operand via lane^16 swap ----
    vi8 bp;
    #pragma unroll
    for (int t = 0; t < 4; ++t) {
      int o1  = packh2(p1[2*t], p1[2*t+1]);   // own S'1 values
      int o2  = packh2(p2[2*t], p2[2*t+1]);   // own S'2 values
      int w1  = swzx16i(o1);                  // partner's S'1
      int w2  = swzx16i(o2);                  // partner's S'2
      bp[t]     = hi ? w2 : o1;
      bp[t + 4] = hi ? o2 : w1;
    }
    const v16h bpm = __builtin_bit_cast(v16h, bp);

    // ---- O[c,j] += V[c,i] * P[i,j], V tile from LDS (ds_load_b128) ----
    const v8h* vbuf = (const v8h*)&vs[it & 1][0];    // 16B chunks, 4 per row
    #pragma unroll
    for (int t = 0; t < 16; ++t) {
      const int c = t * 16 + l16;
      v8h lo8 = vbuf[c * 4 + (hi ? 2 : 0)];
      v8h hh8 = vbuf[c * 4 + (hi ? 3 : 1)];
      v16h av = cat8(lo8, hh8);
      acc[t] = __builtin_amdgcn_wmma_f32_16x16x32_f16(false, av, false, bpm,
                                                      (short)0, acc[t], false, false);
    }
  }

  // ---- finalize: out = x + gamma * O / l ----
  const float g    = gamma[0];
  const float invl = 1.0f / lrun;
  #pragma unroll
  for (int t = 0; t < 16; ++t) {
    #pragma unroll
    for (int r = 0; r < 8; ++r) {
      int c = t * 16 + r + (hi ? 8 : 0);
      size_t idx = ((size_t)b * CC + c) * LL + j;
      out[idx] = x[idx] + g * acc[t][r] * invl;
    }
  }
  // s_endpgm performs an implicit wait-idle for the dangling prefetch
}

// ---------------------------------------------------------------------------
extern "C" void kernel_launch(void* const* d_in, const int* in_sizes, int n_in,
                              void* d_out, int out_size, void* d_ws, size_t ws_size,
                              hipStream_t stream) {
  (void)in_sizes; (void)n_in; (void)out_size; (void)ws_size;
  const float* x     = (const float*)d_in[0];
  const float* Wqkv  = (const float*)d_in[1];
  const float* bqkv  = (const float*)d_in[2];
  const float* gamma = (const float*)d_in[3];
  float* out = (float*)d_out;

  // f16 workspace: qt (4MB) + kt (4MB) + v (16MB) ~= 24MB (fits 192MB L2)
  _Float16* qt = (_Float16*)d_ws;
  _Float16* kt = qt + (size_t)BB * LL * MM;
  _Float16* vv = kt + (size_t)BB * LL * MM;

  // 8 b * 12 o-blocks(32 rows) * 64 l-blocks(64 cols)
  qkv_relu_gemm<<<6144, 256, 0, stream>>>(x, Wqkv, bqkv, qt, kt, vv);
  // 8 b * 32 j-blocks(128 cols)
  flash_attn<<<256, 256, 0, stream>>>(qt, kt, vv, x, gamma, out);
}
